// GAT_60000693125135
// MI455X (gfx1250) — compile-verified, matrix operand order checked by reading, not run on hardware
//
#include <hip/hip_runtime.h>
#include <math.h>

typedef __attribute__((ext_vector_type(16))) _Float16 v16h;
typedef __attribute__((ext_vector_type(8)))  _Float16 v8h;
typedef __attribute__((ext_vector_type(8)))  float    v8f;
typedef _Float16 f16_t;

#define NEG_SLOPE 0.2f

// ---- float <-> order-preserving unsigned encoding (for atomic max on f32) ----
static __device__ __forceinline__ unsigned fenc(float f) {
  unsigned u = __float_as_uint(f);
  return (u & 0x80000000u) ? ~u : (u | 0x80000000u);
}
static __device__ __forceinline__ float fdec(unsigned u) {
  unsigned b = (u & 0x80000000u) ? (u & 0x7FFFFFFFu) : ~u;
  return __uint_as_float(b);
}

// ---------------- utility kernels ----------------
__global__ void k_f2h(const float* __restrict__ in, f16_t* __restrict__ out, int n) {
  int i = blockIdx.x * blockDim.x + threadIdx.x;
  if (i < n) out[i] = (f16_t)in[i];
}
// W[k, c] (row-major KxNcols) -> Wt[c, k] (column-major, i.e. Ncols x K row-major), f32 -> f16
__global__ void k_f2h_transpose(const float* __restrict__ W, f16_t* __restrict__ Wt,
                                int K, int Ncols) {
  int i = blockIdx.x * blockDim.x + threadIdx.x;
  if (i >= K * Ncols) return;
  int k = i / Ncols, c = i - k * Ncols;
  Wt[(size_t)c * K + k] = (f16_t)W[i];
}
__global__ void k_fillf(float* __restrict__ p, float v, int n) {
  int i = blockIdx.x * blockDim.x + threadIdx.x;
  if (i < n) p[i] = v;
}
__global__ void k_fillu(unsigned* __restrict__ p, unsigned v, int n) {
  int i = blockIdx.x * blockDim.x + threadIdx.x;
  if (i < n) p[i] = v;
}

// ---------------- WMMA GEMM: C[M,Ncols] = A[M,K] * Bt[Ncols,K]^T, f16 in, f32 acc ------------
// One wave per 16x16 C tile; blockDim = (32, WY).
// ISA 7.12.2 fragment layout: lane L (m = L&15, half = L>>4) holds, per 32-wide k-step,
// two contiguous 16-byte runs: K in [half*8, half*8+8) and [16+half*8, 16+half*8+8).
// With A row-major and B pre-transposed (lane holds a column of B), both fragments are
// exactly two global_load_b128 each.
__global__ void k_gemm_f16_wmma(const f16_t* __restrict__ A, const f16_t* __restrict__ Bt,
                                float* __restrict__ C, int M, int Ncols, int K) {
  const int lane  = threadIdx.x;
  const int half  = lane >> 4;
  const int lo    = lane & 15;
  const int tileM = blockIdx.x;
  const int tileN = blockIdx.y * blockDim.y + threadIdx.y;

  const f16_t* __restrict__ Ap = A  + (size_t)(tileM * 16 + lo) * K + half * 8;
  const f16_t* __restrict__ Bp = Bt + (size_t)(tileN * 16 + lo) * K + half * 8;

  v8f acc = {};
#pragma unroll 4
  for (int k0 = 0; k0 < K; k0 += 32) {
    v8h a0 = *(const v8h*)(Ap + k0);
    v8h a1 = *(const v8h*)(Ap + k0 + 16);
    v8h b0 = *(const v8h*)(Bp + k0);
    v8h b1 = *(const v8h*)(Bp + k0 + 16);
    v16h a = __builtin_shufflevector(a0, a1, 0, 1, 2, 3, 4, 5, 6, 7,
                                             8, 9, 10, 11, 12, 13, 14, 15);
    v16h b = __builtin_shufflevector(b0, b1, 0, 1, 2, 3, 4, 5, 6, 7,
                                             8, 9, 10, 11, 12, 13, 14, 15);
    acc = __builtin_amdgcn_wmma_f32_16x16x32_f16(
        /*neg_a=*/false, a, /*neg_b=*/false, b,
        /*c_mod=*/(short)0, acc, /*reuse_a=*/false, /*reuse_b=*/false);
  }
  // C/D layout: vgpr r -> row = 8*half + r, col = lane&15
#pragma unroll
  for (int r = 0; r < 8; ++r) {
    int orow = tileM * 16 + (half << 3) + r;
    C[(size_t)orow * Ncols + tileN * 16 + lo] = acc[r];
  }
}

// ---------------- per-node attention dot products ----------------
// h: [N, H, C] flattened; i indexes (node*H + head)
__global__ void k_node_alpha(const float* __restrict__ h, const float* __restrict__ aS,
                             const float* __restrict__ aD, float* __restrict__ oS,
                             float* __restrict__ oD, int NH, int H, int C) {
  int i = blockIdx.x * blockDim.x + threadIdx.x;
  if (i >= NH) return;
  int head = i % H;
  float s = 0.f, d = 0.f;
  for (int c = 0; c < C; ++c) {
    float v = h[i * C + c];
    s += v * aS[head * C + c];
    d += v * aD[head * C + c];
  }
  oS[i] = s;
  oD[i] = d;
}

// ---------------- edge helpers ----------------
static __device__ __forceinline__ void edge_sd(const int* __restrict__ ei, int E_, int eidx,
                                               int& s, int& d) {
  if (eidx < E_) { s = ei[eidx]; d = ei[E_ + eidx]; }
  else           { s = d = eidx - E_; }   // self loop
}

// pass 1: leaky-relu scores + segment max (encoded-uint atomic max)
__global__ void k_edge_scores(const int* __restrict__ ei, int E_, int ET,
                              const float* __restrict__ aS, const float* __restrict__ aD,
                              float* __restrict__ e, unsigned* __restrict__ emax, int H) {
  int t = blockIdx.x * blockDim.x + threadIdx.x;
  if (t >= ET * H) return;
  int eidx = t / H, head = t - eidx * H;
  int s, d; edge_sd(ei, E_, eidx, s, d);
  float v = aS[s * H + head] + aD[d * H + head];
  v = (v > 0.f) ? v : NEG_SLOPE * v;
  e[t] = v;
  atomicMax(&emax[d * H + head], fenc(v));
}

// pass 2: exp(e - max) + segment sum
__global__ void k_edge_softmax(const int* __restrict__ ei, int E_, int ET,
                               float* __restrict__ e, const unsigned* __restrict__ emax,
                               float* __restrict__ denom, int H) {
  int t = blockIdx.x * blockDim.x + threadIdx.x;
  if (t >= ET * H) return;
  int eidx = t / H, head = t - eidx * H;
  int s, d; edge_sd(ei, E_, eidx, s, d); (void)s;
  float p = __expf(e[t] - fdec(emax[d * H + head]));
  e[t] = p;
  atomicAdd(&denom[d * H + head], p);
}

// pass 3: out[dst] += alpha * h[src]; one block per edge, blockDim.x == H*C
__global__ void k_edge_aggregate(const int* __restrict__ ei, int E_,
                                 const float* __restrict__ p, const float* __restrict__ denom,
                                 const float* __restrict__ h, float* __restrict__ out,
                                 int H, int C) {
  int eidx = blockIdx.x;
  int c = threadIdx.x;
  int head = c / C;
  int s, d; edge_sd(ei, E_, eidx, s, d);
  float alpha = p[eidx * H + head] / denom[d * H + head];
  int F = H * C;
  atomicAdd(&out[d * F + c], h[s * F + c] * alpha);
}

// ---------------- layer-1 epilogue: relu(out + b) -> f16 for GEMM2 ----------------
__global__ void k_bias_relu_f16(const float* __restrict__ acc, const float* __restrict__ b,
                                f16_t* __restrict__ hb, int total, int F) {
  int i = blockIdx.x * blockDim.x + threadIdx.x;
  if (i >= total) return;
  float v = acc[i] + b[i % F];
  hb[i] = (f16_t)fmaxf(v, 0.f);
}

// ---------------- layer-2 epilogue: +b2 then log_softmax over 64 classes ----------------
// one wave32 per row, 2 values per lane, shuffle reductions
__global__ void k_bias_logsoftmax(const float* __restrict__ acc, const float* __restrict__ b,
                                  float* __restrict__ out, int N) {
  int row = blockIdx.x;
  int lane = threadIdx.x;               // 0..31
  if (row >= N) return;
  float v0 = acc[row * 64 + lane]      + b[lane];
  float v1 = acc[row * 64 + 32 + lane] + b[32 + lane];
  float mx = fmaxf(v0, v1);
#pragma unroll
  for (int o = 16; o > 0; o >>= 1) mx = fmaxf(mx, __shfl_xor(mx, o, 32));
  float sm = __expf(v0 - mx) + __expf(v1 - mx);
#pragma unroll
  for (int o = 16; o > 0; o >>= 1) sm += __shfl_xor(sm, o, 32);
  float lg = __logf(sm);
  out[row * 64 + lane]      = v0 - mx - lg;
  out[row * 64 + 32 + lane] = v1 - mx - lg;
}

// ---------------- host side ----------------
static inline dim3 g1(int n, int b) { return dim3((unsigned)((n + b - 1) / b)); }

extern "C" void kernel_launch(void* const* d_in, const int* in_sizes, int n_in,
                              void* d_out, int out_size, void* d_ws, size_t ws_size,
                              hipStream_t stream) {
  (void)n_in; (void)out_size; (void)ws_size;
  const float* x     = (const float*)d_in[0];   // [N,256]
  const int*   ei    = (const int*)  d_in[1];   // [2,E]
  const float* W1    = (const float*)d_in[2];   // [256,256]
  const float* aS1   = (const float*)d_in[3];   // [8,32]
  const float* aD1   = (const float*)d_in[4];   // [8,32]
  const float* b1    = (const float*)d_in[5];   // [256]
  const float* W2    = (const float*)d_in[6];   // [256,64]
  const float* aS2   = (const float*)d_in[7];   // [1,64]
  const float* aD2   = (const float*)d_in[8];   // [1,64]
  const float* b2    = (const float*)d_in[9];   // [64]
  float* out = (float*)d_out;                   // [N,64] log-softmax

  const int IN  = 256, F1 = 256, OUT = 64, H1 = 8, C1 = 32;
  const int Nn  = in_sizes[0] / IN;             // 50000
  const int E_  = in_sizes[1] / 2;              // 800000
  const int ET  = E_ + Nn;                      // + self loops

  // ---- carve workspace (~162 MB) ----
  unsigned char* base = (unsigned char*)d_ws;
  size_t off = 0;
  auto carve = [&](size_t bytes) { size_t o = off; off = (off + bytes + 255) & ~(size_t)255; return o; };
  f16_t*    W1t   = (f16_t*)   (base + carve((size_t)IN * F1 * 2));   // [F1, IN]  (col-major W1)
  f16_t*    W2t   = (f16_t*)   (base + carve((size_t)F1 * OUT * 2));  // [OUT, F1] (col-major W2)
  f16_t*    xh    = (f16_t*)   (base + carve((size_t)Nn * IN * 2));   // reused as f16(relu(out1))
  float*    h1    = (float*)   (base + carve((size_t)Nn * F1 * 4));   // front reused as h2 [N,64]
  float*    acc1  = (float*)   (base + carve((size_t)Nn * F1 * 4));   // front reused as acc2 [N,64]
  float*    asrc  = (float*)   (base + carve((size_t)Nn * H1 * 4));
  float*    adst  = (float*)   (base + carve((size_t)Nn * H1 * 4));
  unsigned* emax  = (unsigned*)(base + carve((size_t)Nn * H1 * 4));
  float*    denom = (float*)   (base + carve((size_t)Nn * H1 * 4));
  float*    ebuf  = (float*)   (base + carve((size_t)ET * H1 * 4));
  float*    h2    = h1;     // [N,64]
  float*    acc2  = acc1;   // [N,64]

  const int B = 256;

  // ---- f32 -> f16 operands (weights transposed for column-major B fragments) ----
  k_f2h<<<g1(Nn * IN, B), B, 0, stream>>>(x, xh, Nn * IN);
  k_f2h_transpose<<<g1(IN * F1, B), B, 0, stream>>>(W1, W1t, IN, F1);
  k_f2h_transpose<<<g1(F1 * OUT, B), B, 0, stream>>>(W2, W2t, F1, OUT);

  // ---- layer 1: h1 = x @ W1 (WMMA) ----
  k_gemm_f16_wmma<<<dim3(Nn / 16, F1 / 64), dim3(32, 4), 0, stream>>>(xh, W1t, h1, Nn, F1, IN);

  // attention coefficients per node/head
  k_node_alpha<<<g1(Nn * H1, B), B, 0, stream>>>(h1, aS1, aD1, asrc, adst, Nn * H1, H1, C1);

  // segment softmax + aggregation
  k_fillu<<<g1(Nn * H1, B), B, 0, stream>>>(emax, 0u, Nn * H1);
  k_fillf<<<g1(Nn * H1, B), B, 0, stream>>>(denom, 0.f, Nn * H1);
  k_fillf<<<g1(Nn * F1, B), B, 0, stream>>>(acc1, 0.f, Nn * F1);
  k_edge_scores  <<<g1(ET * H1, B), B, 0, stream>>>(ei, E_, ET, asrc, adst, ebuf, emax, H1);
  k_edge_softmax <<<g1(ET * H1, B), B, 0, stream>>>(ei, E_, ET, ebuf, emax, denom, H1);
  k_edge_aggregate<<<dim3((unsigned)ET), dim3(H1 * C1), 0, stream>>>(ei, E_, ebuf, denom, h1, acc1, H1, C1);

  // relu(acc1 + b1) -> f16, overwriting xh (x no longer needed)
  k_bias_relu_f16<<<g1(Nn * F1, B), B, 0, stream>>>(acc1, b1, xh, Nn * F1, F1);

  // ---- layer 2: h2 = relu(h1out) @ W2 (WMMA) ----
  k_gemm_f16_wmma<<<dim3(Nn / 16, 1), dim3(32, 4), 0, stream>>>(xh, W2t, h2, Nn, OUT, F1);

  k_node_alpha<<<g1(Nn, B), B, 0, stream>>>(h2, aS2, aD2, asrc, adst, Nn, 1, OUT);

  k_fillu<<<g1(Nn, B), B, 0, stream>>>(emax, 0u, Nn);
  k_fillf<<<g1(Nn, B), B, 0, stream>>>(denom, 0.f, Nn);
  k_fillf<<<g1(Nn * OUT, B), B, 0, stream>>>(acc2, 0.f, Nn * OUT);
  k_edge_scores  <<<g1(ET, B), B, 0, stream>>>(ei, E_, ET, asrc, adst, ebuf, emax, 1);
  k_edge_softmax <<<g1(ET, B), B, 0, stream>>>(ei, E_, ET, ebuf, emax, denom, 1);
  k_edge_aggregate<<<dim3((unsigned)ET), dim3(OUT), 0, stream>>>(ei, E_, ebuf, denom, h2, acc2, 1, OUT);

  // +b2, log_softmax over 64 classes -> d_out
  k_bias_logsoftmax<<<dim3((unsigned)Nn), dim3(32), 0, stream>>>(acc2, b2, out, Nn);
}
